// AttentionBlock_36876589203589
// MI455X (gfx1250) — matrix operand order repaired and checked by reading, not compiled, not run on hardware
//
#include <hip/hip_runtime.h>
#include <cstddef>

typedef _Float16 half_t;
typedef __attribute__((ext_vector_type(16))) _Float16 v16h;
typedef __attribute__((ext_vector_type(8)))  float    v8f;

union V16H { v16h v; half_t e[16]; };
union V8F  { v8f  v; float  e[8];  };

#define WMMA_F32_F16(a, b, c) \
  __builtin_amdgcn_wmma_f32_16x16x32_f16(false, (a), false, (b), (short)0, (c), false, false)

static constexpr int B_ = 16, C_ = 512, S_ = 1024, NH = 8, HD = 64, G_ = 32, CG = 16;

// workspace layout, units of half_t (total ~82 MB)
static constexpr size_t HT_OFF = 0;                               // hT [B,S,C]
static constexpr size_t QT_OFF = (size_t)B_ * S_ * C_;            // qT [B,nh,S,hd] (scaled)
static constexpr size_t KT_OFF = QT_OFF * 2;                      // kT [B,nh,S,hd]
static constexpr size_t V_OFF  = QT_OFF * 3;                      // v  [B,nh,hd,S]
static constexpr size_t AT_OFF = QT_OFF * 4;                      // attnT [B,S,C]
static constexpr size_t WQ_OFF = QT_OFF * 5;                      // qkv_w f16 [1536,512]
static constexpr size_t WP_OFF = WQ_OFF + (size_t)3 * C_ * C_;    // proj_w f16 [512,512]

// ---- WMMA fragment helpers (contiguous k order k = hi*16 + j, same for A and B) ----
// A (16x32, MxK): lane m = lane&15; hi = lane>>4; element j -> K = hi*16 + j
__device__ inline v16h load_a_frag(const half_t* __restrict__ rowK0, int hi) {
  V16H a;
#pragma unroll
  for (int j = 0; j < 8; ++j) a.e[j]     = rowK0[hi * 16 + j];
#pragma unroll
  for (int j = 0; j < 8; ++j) a.e[8 + j] = rowK0[hi * 16 + 8 + j];
  return a.v;
}
// B (32x16, KxN): lane n = lane&15; element j -> K = hi*16 + j. Caller bakes hi*16 into ptr.
__device__ inline v16h load_b_frag(const half_t* __restrict__ colK0) {
  V16H b;
#pragma unroll
  for (int j = 0; j < 16; ++j) b.e[j] = colK0[j];
  return b.v;
}

// ============================ GroupNorm ============================
// grid: B*G blocks of 256. Writes normalized activations transposed: hT[b, s, c] (f16)
__global__ __launch_bounds__(256) void gn_kernel(const float* __restrict__ x,
                                                 const float* __restrict__ gamma,
                                                 const float* __restrict__ beta,
                                                 half_t* __restrict__ hT) {
  const int bg = blockIdx.x;
  const int b = bg >> 5, g = bg & 31;
  const int tid = threadIdx.x;
  const float* xp = x + ((size_t)b * C_ + (size_t)g * CG) * S_;

  float s = 0.f, ss = 0.f;
  for (int i = tid; i < CG * S_; i += 256) {
    float v = xp[i];
    s += v; ss += v * v;
  }
  __shared__ float rs[256], rss[256];
  rs[tid] = s; rss[tid] = ss;
  __syncthreads();
  for (int o = 128; o > 0; o >>= 1) {
    if (tid < o) { rs[tid] += rs[tid + o]; rss[tid] += rss[tid + o]; }
    __syncthreads();
  }
  const float invN = 1.0f / (CG * S_);
  const float mean = rs[0] * invN;
  const float var  = rss[0] * invN - mean * mean;
  const float rstd = rsqrtf(var + 1e-5f);

  for (int i = tid; i < CG * S_; i += 256) {
    int c = g * CG + (i >> 10);
    int sp = i & (S_ - 1);
    float v = (xp[i] - mean) * rstd * gamma[c] + beta[c];
    hT[((size_t)b * S_ + sp) * C_ + c] = (half_t)v;
  }
}

// ============================ weight fp32 -> f16 ============================
__global__ __launch_bounds__(256) void cvt_kernel(const float* __restrict__ qkvw,
                                                  const float* __restrict__ projw,
                                                  half_t* __restrict__ wq,
                                                  half_t* __restrict__ wp) {
  int i = blockIdx.x * 256 + threadIdx.x;
  if (i < 3 * C_ * C_) wq[i] = (half_t)qkvw[i];
  if (i < C_ * C_)     wp[i] = (half_t)projw[i];
}

// ============================ QKV GEMM ============================
// Out[b,o,s] = sum_c W[o,c] * h[b,c,s] + bias[o]; per-wave 32(o) x 64(s) tile, K=512.
// tasks = 16 * 48 * 16 = 12288 waves -> 1536 blocks of 256.
__global__ __launch_bounds__(256) void qkv_kernel(const half_t* __restrict__ hT,
                                                  const half_t* __restrict__ wq,
                                                  const float* __restrict__ qkvb,
                                                  half_t* __restrict__ qT,
                                                  half_t* __restrict__ kT,
                                                  half_t* __restrict__ vv) {
  const int wave = blockIdx.x * 8 + (threadIdx.x >> 5);
  const int nchunk = wave & 15;
  const int t = wave >> 4;
  const int mtile = t % 48;
  const int b = t / 48;
  const int lane = threadIdx.x & 31;
  const int hi = lane >> 4, ln = lane & 15;
  const int obase = mtile * 32, sbase = nchunk * 64;

  const half_t* arow0 = wq + (size_t)(obase + ln) * C_;
  const half_t* arow1 = wq + (size_t)(obase + 16 + ln) * C_;
  const half_t* brow0 = hT + ((size_t)b * S_ + sbase + ln) * C_ + hi * 16;

  V8F acc[2][4] = {};
  for (int k = 0; k < C_; k += 32) {
    __builtin_prefetch(arow0 + k + 64, 0, 1);
    v16h a0 = load_a_frag(arow0 + k, hi);
    v16h a1 = load_a_frag(arow1 + k, hi);
#pragma unroll
    for (int nt = 0; nt < 4; ++nt) {
      v16h bf = load_b_frag(brow0 + (size_t)nt * 16 * C_ + k);
      acc[0][nt].v = WMMA_F32_F16(a0, bf, acc[0][nt].v);
      acc[1][nt].v = WMMA_F32_F16(a1, bf, acc[1][nt].v);
    }
  }

  // obase is a multiple of 32, q/k/v boundaries at 512/1024 -> whole 32-tile in one segment
  if (obase < C_) {            // ---- Q (1/sqrt(hd) folded in) ----
#pragma unroll
    for (int mt = 0; mt < 2; ++mt)
#pragma unroll
      for (int nt = 0; nt < 4; ++nt)
#pragma unroll
        for (int r = 0; r < 8; ++r) {
          int o = obase + mt * 16 + r + hi * 8;
          int s = sbase + nt * 16 + ln;
          float val = acc[mt][nt].e[r] + qkvb[o];
          int head = o >> 6, d = o & 63;
          qT[(((size_t)b * NH + head) * S_ + s) * HD + d] = (half_t)(val * 0.125f);
        }
  } else if (obase < 2 * C_) { // ---- K ----
#pragma unroll
    for (int mt = 0; mt < 2; ++mt)
#pragma unroll
      for (int nt = 0; nt < 4; ++nt)
#pragma unroll
        for (int r = 0; r < 8; ++r) {
          int o = obase + mt * 16 + r + hi * 8;
          int s = sbase + nt * 16 + ln;
          float val = acc[mt][nt].e[r] + qkvb[o];
          int o2 = o - C_;
          int head = o2 >> 6, d = o2 & 63;
          kT[(((size_t)b * NH + head) * S_ + s) * HD + d] = (half_t)val;
        }
  } else {                     // ---- V (kept [hd, S]) ----
#pragma unroll
    for (int mt = 0; mt < 2; ++mt)
#pragma unroll
      for (int nt = 0; nt < 4; ++nt)
#pragma unroll
        for (int r = 0; r < 8; ++r) {
          int o = obase + mt * 16 + r + hi * 8;
          int s = sbase + nt * 16 + ln;
          float val = acc[mt][nt].e[r] + qkvb[o];
          int o3 = o - 2 * C_;
          int head = o3 >> 6, d = o3 & 63;
          vv[(((size_t)b * NH + head) * HD + d) * S_ + s] = (half_t)val;
        }
  }
}

// ============================ Attention ============================
// 4 waves (128 threads) per (b, head, 16-row query block) sharing one 64KB score buffer.
// Waves split t for the score GEMM, split K for P.V with an LDS-atomic reduction.
// grid: 16*8*64 = 8192 blocks of 128. LDS = 68KB -> 4 blocks/WGP -> 16 waves/WGP.
__global__ __launch_bounds__(128) void attn_kernel(const half_t* __restrict__ qT,
                                                   const half_t* __restrict__ kT,
                                                   const half_t* __restrict__ vv,
                                                   half_t* __restrict__ attnT) {
  __shared__ float sc[16 * 1024];   // 64 KB score block
  __shared__ float obuf[16 * 64];   // 4 KB split-K reduction buffer

  const int blk = blockIdx.x;
  const int stile = blk & 63;
  const int bh = blk >> 6;          // b*NH + head
  const int b = bh >> 3, head = bh & 7;
  const int tid = threadIdx.x;
  const int wv = tid >> 5;
  const int lane = tid & 31;
  const int hi = lane >> 4, ln = lane & 15;
  const int sbase = stile * 16;

  // Q block A-fragments (d = 0..31, 32..63); same for all 4 waves
  const half_t* qrow = qT + ((size_t)bh * S_ + sbase + ln) * HD;
  v16h aq0 = load_a_frag(qrow, hi);
  v16h aq1 = load_a_frag(qrow + 32, hi);

  // scores: wave wv covers t-tiles [wv*16, wv*16+16)
  for (int tt = wv * 16; tt < wv * 16 + 16; ++tt) {
    const half_t* krow = kT + ((size_t)bh * S_ + tt * 16 + ln) * HD + hi * 16;
    V8F c = {};
    c.v = WMMA_F32_F16(aq0, load_b_frag(krow), c.v);
    c.v = WMMA_F32_F16(aq1, load_b_frag(krow + 32), c.v);
#pragma unroll
    for (int r = 0; r < 8; ++r)
      sc[(r + hi * 8) * 1024 + tt * 16 + ln] = c.e[r];
  }
  for (int i = tid; i < 16 * 64; i += 128) obuf[i] = 0.f;
  __syncthreads();

  // softmax: 8 threads per row (groups stay inside a wave -> shuffle reduce)
  {
    const int row = tid >> 3, sub = tid & 7;
    float* rp = sc + row * 1024;
    float m = -1e30f;
    for (int i = sub; i < 1024; i += 8) m = fmaxf(m, rp[i]);
    m = fmaxf(m, __shfl_xor(m, 1, 32));
    m = fmaxf(m, __shfl_xor(m, 2, 32));
    m = fmaxf(m, __shfl_xor(m, 4, 32));
    float ssum = 0.f;
    for (int i = sub; i < 1024; i += 8) {
      float e = __expf(rp[i] - m);
      rp[i] = e;
      ssum += e;
    }
    ssum += __shfl_xor(ssum, 1, 32);
    ssum += __shfl_xor(ssum, 2, 32);
    ssum += __shfl_xor(ssum, 4, 32);
    float inv = 1.0f / ssum;
    for (int i = sub; i < 1024; i += 8) rp[i] *= inv;
  }
  __syncthreads();

  // out[s, d] = P . V, split-K: wave wv covers K-chunks [wv*8, wv*8+8)
  V8F acc[4] = {};
  for (int tc = wv * 8; tc < wv * 8 + 8; ++tc) {
    V16H a;
    const float* ar = sc + ln * 1024 + tc * 32;
#pragma unroll
    for (int j = 0; j < 8; ++j) a.e[j]     = (half_t)ar[hi * 16 + j];
#pragma unroll
    for (int j = 0; j < 8; ++j) a.e[8 + j] = (half_t)ar[hi * 16 + 8 + j];
#pragma unroll
    for (int nt = 0; nt < 4; ++nt) {
      const half_t* vr = vv + (((size_t)bh * HD) + nt * 16 + ln) * S_ + tc * 32 + hi * 16;
      acc[nt].v = WMMA_F32_F16(a.v, load_b_frag(vr), acc[nt].v);
    }
  }
  // cross-wave reduction via LDS float atomics (ds_add_f32)
#pragma unroll
  for (int nt = 0; nt < 4; ++nt)
#pragma unroll
    for (int r = 0; r < 8; ++r)
      atomicAdd(&obuf[(r + hi * 8) * 64 + nt * 16 + ln], acc[nt].e[r]);
  __syncthreads();

  // store transposed for proj GEMM: attnT[b, s, c], c = head*64 + d
#pragma unroll
  for (int e = 0; e < 8; ++e) {
    int idx = tid * 8 + e;
    int m = idx >> 6, d = idx & 63;
    attnT[((size_t)b * S_ + sbase + m) * C_ + head * HD + d] = (half_t)obuf[idx];
  }
}

// ============================ Proj GEMM + bias + residual ============================
// out[b,o,s] = sum_c Wp[o,c]*attn[b,c,s] + pb[o] + x[b,o,s]; 32x64 tiles.
// tasks = 16 * 16 * 16 = 4096 waves -> 512 blocks of 256.
__global__ __launch_bounds__(256) void proj_kernel(const half_t* __restrict__ attnT,
                                                   const half_t* __restrict__ wp,
                                                   const float* __restrict__ pb,
                                                   const float* __restrict__ x,
                                                   float* __restrict__ out) {
  const int wave = blockIdx.x * 8 + (threadIdx.x >> 5);
  const int nchunk = wave & 15;
  const int t = wave >> 4;
  const int mtile = t & 15;
  const int b = t >> 4;
  const int lane = threadIdx.x & 31;
  const int hi = lane >> 4, ln = lane & 15;
  const int obase = mtile * 32, sbase = nchunk * 64;

  const half_t* arow0 = wp + (size_t)(obase + ln) * C_;
  const half_t* arow1 = wp + (size_t)(obase + 16 + ln) * C_;
  const half_t* brow0 = attnT + ((size_t)b * S_ + sbase + ln) * C_ + hi * 16;

  V8F acc[2][4] = {};
  for (int k = 0; k < C_; k += 32) {
    __builtin_prefetch(arow0 + k + 64, 0, 1);
    v16h a0 = load_a_frag(arow0 + k, hi);
    v16h a1 = load_a_frag(arow1 + k, hi);
#pragma unroll
    for (int nt = 0; nt < 4; ++nt) {
      v16h bf = load_b_frag(brow0 + (size_t)nt * 16 * C_ + k);
      acc[0][nt].v = WMMA_F32_F16(a0, bf, acc[0][nt].v);
      acc[1][nt].v = WMMA_F32_F16(a1, bf, acc[1][nt].v);
    }
  }
#pragma unroll
  for (int mt = 0; mt < 2; ++mt)
#pragma unroll
    for (int nt = 0; nt < 4; ++nt)
#pragma unroll
      for (int r = 0; r < 8; ++r) {
        int o = obase + mt * 16 + r + hi * 8;
        int s = sbase + nt * 16 + ln;
        size_t idx = ((size_t)b * C_ + o) * S_ + s;
        out[idx] = acc[mt][nt].e[r] + pb[o] + x[idx];
      }
}

// ============================ host launch ============================
extern "C" void kernel_launch(void* const* d_in, const int* in_sizes, int n_in,
                              void* d_out, int out_size, void* d_ws, size_t ws_size,
                              hipStream_t stream) {
  const float* x     = (const float*)d_in[0];
  const float* nw    = (const float*)d_in[1];
  const float* nb    = (const float*)d_in[2];
  const float* qkvw  = (const float*)d_in[3];
  const float* qkvb  = (const float*)d_in[4];
  const float* projw = (const float*)d_in[5];
  const float* projb = (const float*)d_in[6];
  float* out = (float*)d_out;

  half_t* ws = (half_t*)d_ws;
  half_t* hT = ws + HT_OFF;
  half_t* qT = ws + QT_OFF;
  half_t* kT = ws + KT_OFF;
  half_t* vv = ws + V_OFF;
  half_t* aT = ws + AT_OFF;
  half_t* wq = ws + WQ_OFF;
  half_t* wp = ws + WP_OFF;

  gn_kernel<<<B_ * G_, 256, 0, stream>>>(x, nw, nb, hT);
  cvt_kernel<<<(3 * C_ * C_ + 255) / 256, 256, 0, stream>>>(qkvw, projw, wq, wp);
  qkv_kernel<<<(16 * 48 * 16) / 8, 256, 0, stream>>>(hT, wq, qkvb, qT, kT, vv);
  attn_kernel<<<B_ * NH * (S_ / 16), 128, 0, stream>>>(qT, kT, vv, aT);
  proj_kernel<<<(16 * 16 * 16) / 8, 256, 0, stream>>>(aT, wp, projb, x, out);
}
